// RecursiveQuantumTransformerLayer_41205916238248
// MI455X (gfx1250) — matrix operand order repaired and checked by reading, not compiled, not run on hardware
//
#include <hip/hip_runtime.h>

// ==================== problem constants ====================
#define B_ 4
#define S_ 1024
#define E_ 1024
#define F_ 4096
#define H_ 16
#define D_ 64
#define M_ (B_ * S_)            // 4096 token rows
#define ME_ ((long)M_ * E_)     // 4,194,304 elements

typedef __attribute__((ext_vector_type(16))) __bf16 v16bf;
typedef __attribute__((ext_vector_type(8)))  float  v8f;

#define WMMA_BF16(a, b, c) \
  __builtin_amdgcn_wmma_f32_16x16x32_bf16(false, (a), false, (b), (short)0, (c), false, false)

enum { EPI_BIAS = 0, EPI_GELU = 1, EPI_ADD_SIGMOID = 2, EPI_RES_SCALE = 3 };

template <bool BF16B> struct BSel { using T = float; };
template <>           struct BSel<true> { using T = __bf16; };

// ==================== WMMA GEMM (bf16 compute, f32 accum/out) ====================
// C[M,N] = epi(A[M,K] @ B[K,N] + bias). Block = 128x128, K-step 32, 256 thr (8 waves),
// each wave owns a 32x64 sub-tile = 2x4 WMMA 16x16 accumulators.
// Double-buffered LDS pipeline: global loads for tile t+1 overlap the WMMAs of tile t.
// BF16B: B operand is pre-converted bf16 (weights); else f32 converted on the fly.
// All dims are multiples of 128 in this workload (no bounds checks needed).
template <int EPI, bool TRANSB, bool BF16B>
__global__ __launch_bounds__(256) void gemm_wmma_k(
    const float* __restrict__ A, const void* __restrict__ Bv,
    const float* __restrict__ bias, const float* __restrict__ res,
    const float* __restrict__ scale_p, float* __restrict__ C,
    int K, int lda, int ldb, int ldc, long sA, long sB, long sC)
{
  using BT = typename BSel<BF16B>::T;
  const BT* __restrict__ Bm = (const BT*)Bv;

  __shared__ __bf16 As[2][128][40];   // [buf][m][k], padded
  __shared__ __bf16 Bs[2][32][136];   // [buf][k][n], padded
  const int tid = threadIdx.x, lane = tid & 31, w = tid >> 5;
  const int wr = w >> 1, wc = w & 1;
  const int bm = blockIdx.x * 128, bn = blockIdx.y * 128;
  A  += (long)blockIdx.z * sA;
  Bm += (long)blockIdx.z * sB;
  C  += (long)blockIdx.z * sC;

  v8f zero = {};
  v8f acc[2][4];
  for (int i = 0; i < 2; ++i) for (int j = 0; j < 4; ++j) acc[i][j] = zero;

  const int arow = tid >> 1, acb = (tid & 1) * 16;   // A tile: 128 rows x 32 k
  const int brow = tid >> 3, bcb = (tid & 7) * 16;   // B tile: 32 k x 128 cols
  const int mrow = lane & 15;
  const int kb8  = (lane >> 4) ? 8 : 0;              // unused: A fragments also take K base kb16 (k = 16h + s)
  const int kb16 = (lane >> 4) ? 16 : 0;             // B-fragment K base

  float aR[16];
  BT    bR[16];

  // ---- prologue: fetch + stage tile 0 ----
#pragma unroll
  for (int j = 0; j < 16; ++j)
    aR[j] = A[(long)(bm + arow) * lda + (acb + j)];
#pragma unroll
  for (int j = 0; j < 16; ++j)
    bR[j] = TRANSB ? Bm[(long)(bn + bcb + j) * ldb + brow]
                   : Bm[(long)brow * ldb + (bn + bcb + j)];
#pragma unroll
  for (int j = 0; j < 16; ++j) As[0][arow][acb + j] = (__bf16)aR[j];
#pragma unroll
  for (int j = 0; j < 16; ++j) Bs[0][brow][bcb + j] = (__bf16)bR[j];
  __syncthreads();

  const int nT = K >> 5;
  int buf = 0;
  for (int t = 0; t < nT; ++t) {
    // ---- issue global loads for tile t+1 (hidden under the WMMAs below) ----
    if (t + 1 < nT) {
      const int k0 = (t + 1) << 5;
#pragma unroll
      for (int j = 0; j < 16; ++j)
        aR[j] = A[(long)(bm + arow) * lda + (k0 + acb + j)];
#pragma unroll
      for (int j = 0; j < 16; ++j)
        bR[j] = TRANSB ? Bm[(long)(bn + bcb + j) * ldb + (k0 + brow)]
                       : Bm[(long)(k0 + brow) * ldb + (bn + bcb + j)];
    }

    // ---- fragments + 8 WMMAs from LDS buffer `buf` ----
    v16bf afr[2], bfr[4];
#pragma unroll
    for (int mi = 0; mi < 2; ++mi)
#pragma unroll
      for (int j = 0; j < 16; ++j) {
        int vq = j >> 1;
        int kk = kb16 + j;
        afr[mi][j] = As[buf][wr * 32 + mi * 16 + mrow][kk];
      }
#pragma unroll
    for (int ni = 0; ni < 4; ++ni)
#pragma unroll
      for (int j = 0; j < 16; ++j)
        bfr[ni][j] = Bs[buf][kb16 + j][wc * 64 + ni * 16 + (lane & 15)];

#pragma unroll
    for (int mi = 0; mi < 2; ++mi)
#pragma unroll
      for (int ni = 0; ni < 4; ++ni)
        acc[mi][ni] = WMMA_BF16(afr[mi], bfr[ni], acc[mi][ni]);

    // ---- stage tile t+1 into the other buffer ----
    if (t + 1 < nT) {
#pragma unroll
      for (int j = 0; j < 16; ++j) As[buf ^ 1][arow][acb + j] = (__bf16)aR[j];
#pragma unroll
      for (int j = 0; j < 16; ++j) Bs[buf ^ 1][brow][bcb + j] = (__bf16)bR[j];
    }
    __syncthreads();
    buf ^= 1;
  }

  // ---- fused epilogue ----
  const int roff = (lane >> 4) ? 8 : 0;
  const int c0 = lane & 15;
  const float sc = (EPI == EPI_RES_SCALE) ? scale_p[0] : 0.f;
#pragma unroll
  for (int mi = 0; mi < 2; ++mi)
#pragma unroll
    for (int ni = 0; ni < 4; ++ni) {
      const int colg = bn + wc * 64 + ni * 16 + c0;
      const float bv = bias ? bias[colg] : 0.f;
#pragma unroll
      for (int r = 0; r < 8; ++r) {
        const long row = bm + wr * 32 + mi * 16 + r + roff;
        const long idx = row * (long)ldc + colg;
        float v = acc[mi][ni][r] + bv;
        if (EPI == EPI_BIAS)        C[idx] = v;
        else if (EPI == EPI_GELU)   C[idx] = 0.5f * v * (1.f + erff(v * 0.70710678f));
        else if (EPI == EPI_ADD_SIGMOID) { float t2 = C[idx] + v; C[idx] = 1.f / (1.f + __expf(-t2)); }
        else                        C[idx] = res[idx] + sc * v;   // EPI_RES_SCALE
      }
    }
}

// ==================== flash attention (per b,h; q-tile 128) ====================
// q,k,v,o layout: [B*S, E] with head offset h*64; scale = 1/sqrt(64) = 0.125
__global__ __launch_bounds__(256) void flash_attn_k(
    const float* __restrict__ Q, const float* __restrict__ Kg,
    const float* __restrict__ Vg, float* __restrict__ O)
{
  __shared__ __bf16 Ks[64][72];
  __shared__ __bf16 Vs[64][72];
  __shared__ __bf16 Ps[128][72];   // Q staging, then per-wave P tiles
  const int tid = threadIdx.x, lane = tid & 31, w = tid >> 5;
  const int qb0 = blockIdx.x * 128;
  const long base = ((long)blockIdx.z * S_) * E_ + blockIdx.y * D_;

  { // stage Q tile (rows are wave-private: rows [w*16, w*16+16))
    const int r = tid >> 1, cb = (tid & 1) * 32;
#pragma unroll
    for (int j = 0; j < 32; ++j)
      Ps[r][cb + j] = (__bf16)Q[base + (long)(qb0 + r) * E_ + cb + j];
  }
  __syncthreads();

  const int mrow = lane & 15;
  const int kb8 = (lane >> 4) ? 8 : 0, kb16 = (lane >> 4) ? 16 : 0;
  v16bf aq[2];                       // Q fragments stay in registers
#pragma unroll
  for (int c = 0; c < 2; ++c)
#pragma unroll
    for (int j = 0; j < 16; ++j) {
      int vq = j >> 1;
      int kk = c * 32 + kb16 + j;
      aq[c][j] = Ps[w * 16 + mrow][kk];
    }

  v8f zero = {};
  v8f o[4]; for (int i = 0; i < 4; ++i) o[i] = zero;
  float mrun[8], lrun[8];
#pragma unroll
  for (int r = 0; r < 8; ++r) { mrun[r] = -3.0e38f; lrun[r] = 0.f; }
  __syncthreads();

  const int kr = tid >> 2, cb2 = (tid & 3) * 16;
  const int proff = (lane >> 4) ? 8 : 0, pc = lane & 15;

  for (int kt = 0; kt < S_ / 64; ++kt) {
#pragma unroll
    for (int j = 0; j < 16; ++j) {
      Ks[kr][cb2 + j] = (__bf16)Kg[base + (long)(kt * 64 + kr) * E_ + cb2 + j];
      Vs[kr][cb2 + j] = (__bf16)Vg[base + (long)(kt * 64 + kr) * E_ + cb2 + j];
    }
    __syncthreads();

    // S = Q K^T over this 64-key tile (per wave: 16 q-rows x 64 keys)
    v8f s[4];
#pragma unroll
    for (int ni = 0; ni < 4; ++ni) {
      v16bf bk0, bk1;
#pragma unroll
      for (int j = 0; j < 16; ++j) {
        bk0[j] = Ks[ni * 16 + (lane & 15)][kb16 + j];         // B(d,key)=K[key][d]
        bk1[j] = Ks[ni * 16 + (lane & 15)][32 + kb16 + j];
      }
      v8f t = zero;
      t = WMMA_BF16(aq[0], bk0, t);
      t = WMMA_BF16(aq[1], bk1, t);
      s[ni] = t;
    }

    // online softmax: per-row stats via 16-lane shuffle reductions
#pragma unroll
    for (int r = 0; r < 8; ++r) {
      float mx = -3.0e38f;
#pragma unroll
      for (int ni = 0; ni < 4; ++ni) mx = fmaxf(mx, s[ni][r]);
#pragma unroll
      for (int off = 8; off >= 1; off >>= 1) mx = fmaxf(mx, __shfl_xor(mx, off, 32));
      mx *= 0.125f;                                   // scale > 0: max commutes
      const float mn = fmaxf(mrun[r], mx);
      const float corr = __expf(mrun[r] - mn);
      mrun[r] = mn;
      float psum = 0.f;
#pragma unroll
      for (int ni = 0; ni < 4; ++ni) {
        float p = __expf(s[ni][r] * 0.125f - mn);
        psum += p;
        Ps[w * 16 + r + proff][ni * 16 + pc] = (__bf16)p;   // wave-private region
      }
#pragma unroll
      for (int off = 8; off >= 1; off >>= 1) psum += __shfl_xor(psum, off, 32);
      lrun[r] = lrun[r] * corr + psum;
#pragma unroll
      for (int ni = 0; ni < 4; ++ni) o[ni][r] *= corr;
    }

    // O += P @ V  (A-fragments gathered back from wave-private Ps rows)
#pragma unroll
    for (int c = 0; c < 2; ++c) {
      v16bf ap;
#pragma unroll
      for (int j = 0; j < 16; ++j) {
        int vq = j >> 1;
        int kk = c * 32 + kb16 + j;
        ap[j] = Ps[w * 16 + mrow][kk];
      }
#pragma unroll
      for (int ni = 0; ni < 4; ++ni) {
        v16bf bv;
#pragma unroll
        for (int j = 0; j < 16; ++j)
          bv[j] = Vs[c * 32 + kb16 + j][ni * 16 + (lane & 15)];
        o[ni] = WMMA_BF16(ap, bv, o[ni]);
      }
    }
    __syncthreads();
  }

#pragma unroll
  for (int ni = 0; ni < 4; ++ni)
#pragma unroll
    for (int r = 0; r < 8; ++r) {
      const long srow = qb0 + w * 16 + r + proff;
      O[base + srow * E_ + ni * 16 + (lane & 15)] = o[ni][r] / lrun[r];
    }
}

// ==================== block reductions (256 threads = 8 waves) ====================
__device__ __forceinline__ float blk_sum(float v, float* sh) {
#pragma unroll
  for (int off = 16; off >= 1; off >>= 1) v += __shfl_xor(v, off, 32);
  if ((threadIdx.x & 31) == 0) sh[threadIdx.x >> 5] = v;
  __syncthreads();
  float r = 0.f;
#pragma unroll
  for (int i = 0; i < 8; ++i) r += sh[i];
  __syncthreads();
  return r;
}
__device__ __forceinline__ float blk_max(float v, float* sh) {
#pragma unroll
  for (int off = 16; off >= 1; off >>= 1) v = fmaxf(v, __shfl_xor(v, off, 32));
  if ((threadIdx.x & 31) == 0) sh[threadIdx.x >> 5] = v;
  __syncthreads();
  float r = -3.0e38f;
#pragma unroll
  for (int i = 0; i < 8; ++i) r = fmaxf(r, sh[i]);
  __syncthreads();
  return r;
}

// ==================== rowwise kernels ====================
__global__ __launch_bounds__(256) void layernorm_k(
    const float* __restrict__ X, const float* __restrict__ g,
    const float* __restrict__ be, float* __restrict__ Y)
{
  __shared__ float sh[8];
  const long row = blockIdx.x;
  const float* xr = X + row * (long)E_;
  float v[4], s = 0.f, q = 0.f;
#pragma unroll
  for (int i = 0; i < 4; ++i) { v[i] = xr[threadIdx.x + i * 256]; s += v[i]; q += v[i] * v[i]; }
  s = blk_sum(s, sh);
  q = blk_sum(q, sh);
  const float mean = s * (1.f / E_);
  const float var  = q * (1.f / E_) - mean * mean;
  const float rstd = rsqrtf(var + 1e-5f);
#pragma unroll
  for (int i = 0; i < 4; ++i) {
    const int c = threadIdx.x + i * 256;
    Y[row * (long)E_ + c] = (v[i] - mean) * rstd * g[c] + be[c];
  }
}

__global__ __launch_bounds__(256) void softmax_rows_k(float* __restrict__ Sm,
                                                      const float* __restrict__ tp)
{
  __shared__ float sh[8];
  const long row = blockIdx.x;
  float* xr = Sm + row * (long)S_;
  const float sc = 1.f / (32.f * tp[0]);     // /sqrt(E) then /temperature
  float v[4], mx = -3.0e38f;
#pragma unroll
  for (int i = 0; i < 4; ++i) { v[i] = xr[threadIdx.x + i * 256] * sc; mx = fmaxf(mx, v[i]); }
  mx = blk_max(mx, sh);
  float s = 0.f;
#pragma unroll
  for (int i = 0; i < 4; ++i) { v[i] = __expf(v[i] - mx); s += v[i]; }
  s = blk_sum(s, sh);
  const float inv = 1.f / s;
#pragma unroll
  for (int i = 0; i < 4; ++i) xr[threadIdx.x + i * 256] = v[i] * inv;
}

__global__ __launch_bounds__(256) void ent_bal_k(
    const float* __restrict__ iface, const float* __restrict__ eew,
    const float* __restrict__ eeb, float* __restrict__ ent, float* __restrict__ bal)
{
  __shared__ float sh[8];
  const long row = blockIdx.x;
  const float* xr = iface + row * (long)E_;
  float v[4], mx = -3.0e38f;
#pragma unroll
  for (int i = 0; i < 4; ++i) { v[i] = xr[threadIdx.x + i * 256]; mx = fmaxf(mx, v[i]); }
  mx = blk_max(mx, sh);
  float s1 = 0.f;
#pragma unroll
  for (int i = 0; i < 4; ++i) s1 += __expf(v[i] - mx);
  s1 = blk_sum(s1, sh);
  const float lse = mx + __logf(s1);
  float s2 = 0.f, dot = 0.f;
#pragma unroll
  for (int i = 0; i < 4; ++i) {
    const float lp = v[i] - lse;
    s2 += __expf(lp) * lp;
    dot += v[i] * eew[threadIdx.x + i * 256];
  }
  s2 = blk_sum(s2, sh);
  dot = blk_sum(dot, sh);
  if (threadIdx.x == 0) {
    ent[row] = -s2;                                        // entropy
    bal[row] = 1.f / (1.f + __expf(-(dot + eeb[0])));      // sigmoid balance
  }
}

// ==================== elementwise kernels ====================
__global__ __launch_bounds__(256) void cvt_bf16_k(const float* __restrict__ a,
                                                  __bf16* __restrict__ o)
{
  const long i = (long)blockIdx.x * 1024 + threadIdx.x;
#pragma unroll
  for (int j = 0; j < 4; ++j) o[i + j * 256] = (__bf16)a[i + j * 256];
}
__global__ __launch_bounds__(256) void init_xk_k(const float* __restrict__ x,
                                                 float* __restrict__ xc, float* __restrict__ ks)
{
  const long i = (long)blockIdx.x * 1024 + threadIdx.x;
#pragma unroll
  for (int j = 0; j < 4; ++j) { float v = x[i + j * 256]; xc[i + j * 256] = v; ks[i + j * 256] = v; }
}
__global__ __launch_bounds__(256) void iface_k(float* __restrict__ cls,
                                               const float* __restrict__ gate,
                                               const float* __restrict__ tr)
{
  const long i = (long)blockIdx.x * 1024 + threadIdx.x;
#pragma unroll
  for (int j = 0; j < 4; ++j) cls[i + j * 256] += gate[i + j * 256] * tr[i + j * 256];
}
__global__ __launch_bounds__(256) void reg_k(float* __restrict__ x, const float* __restrict__ ifc,
    const float* __restrict__ ne, const float* __restrict__ ent, const float* __restrict__ bal,
    const float* __restrict__ eta_p)
{
  const long row = blockIdx.x;
  const float f = eta_p[0] * bal[row] / (ent[row] + 1e-6f);
  const long b0 = row * (long)E_ + threadIdx.x;
#pragma unroll
  for (int j = 0; j < 4; ++j) x[b0 + j * 256] += ifc[b0 + j * 256] + f * ne[b0 + j * 256];
}
__global__ __launch_bounds__(256) void ks_upd_k(float* __restrict__ ks, const float* __restrict__ xn)
{
  const long i = (long)blockIdx.x * 1024 + threadIdx.x;
#pragma unroll
  for (int j = 0; j < 4; ++j) ks[i + j * 256] += 0.1f * (xn[i + j * 256] - ks[i + j * 256]);
}
__global__ __launch_bounds__(256) void copy_k(const float* __restrict__ a, float* __restrict__ o)
{
  const long i = (long)blockIdx.x * 1024 + threadIdx.x;
#pragma unroll
  for (int j = 0; j < 4; ++j) o[i + j * 256] = a[i + j * 256];
}

// ==================== host orchestration ====================
extern "C" void kernel_launch(void* const* d_in, const int* in_sizes, int n_in,
                              void* d_out, int out_size, void* d_ws, size_t ws_size,
                              hipStream_t stream)
{
  const float* x_in  = (const float*)d_in[0];
  const float* sr_w  = (const float*)d_in[1];
  const float* sr_b  = (const float*)d_in[2];
  const float* q_w   = (const float*)d_in[3];
  const float* q_b   = (const float*)d_in[4];
  const float* k_w   = (const float*)d_in[5];
  const float* k_b   = (const float*)d_in[6];
  const float* v_w   = (const float*)d_in[7];
  const float* v_b   = (const float*)d_in[8];
  const float* o_w   = (const float*)d_in[9];
  const float* o_b   = (const float*)d_in[10];
  const float* refg  = (const float*)d_in[11];
  const float* c_w   = (const float*)d_in[12];
  const float* c_b   = (const float*)d_in[13];
  const float* qp_w  = (const float*)d_in[14];
  const float* qp_b  = (const float*)d_in[15];
  const float* ig_w  = (const float*)d_in[16];
  const float* ig_b  = (const float*)d_in[17];
  const float* temp  = (const float*)d_in[18];
  const float* kp_w  = (const float*)d_in[19];
  const float* kp_b  = (const float*)d_in[20];
  const float* ee_w  = (const float*)d_in[21];
  const float* ee_b  = (const float*)d_in[22];
  const float* eta   = (const float*)d_in[23];
  const float* f1_w  = (const float*)d_in[24];
  const float* f1_b  = (const float*)d_in[25];
  const float* f2_w  = (const float*)d_in[26];
  const float* f2_b  = (const float*)d_in[27];
  const float* ln1_g = (const float*)d_in[28];
  const float* ln1_b = (const float*)d_in[29];
  const float* ln2_g = (const float*)d_in[30];
  const float* ln2_b = (const float*)d_in[31];
  const float* ln3_g = (const float*)d_in[32];
  const float* ln3_b = (const float*)d_in[33];
  const float* obs_w = (const float*)d_in[34];
  const float* obs_b = (const float*)d_in[35];
  const float* obsg  = (const float*)d_in[36];

  // workspace layout (floats); f32 part = 16*ME_ + 2*M_, then 19*E*E bf16 weights
  // total ~= 268.5 MB + 38 MB
  float* ws  = (float*)d_ws;
  float* x0  = ws;
  float* x1  = ws + 1 * ME_;
  float* ksb = ws + 2 * ME_;
  float* xn  = ws + 3 * ME_;     // ln1 out; later reused for neg_entropy
  float* xr  = ws + 4 * ME_;     // self-ref out; later reused as ln2 out
  float* qb  = ws + 5 * ME_;     // q; later c_proj
  float* kb  = ws + 6 * ME_;     // k; later q_proj
  float* vb  = ws + 7 * ME_;     // v; later transfer
  float* ao  = ws + 8 * ME_;     // attn out; later gate
  float* qu  = ws + 9 * ME_;     // quantum
  float* cl  = ws + 10 * ME_;    // classical / iface (in place)
  float* hb  = ws + 11 * ME_;    // FFN hidden [M,F] (4*ME_)
  float* sim = ws + 15 * ME_;    // [B,S,S] (== ME_)
  float* ent = ws + 16 * ME_;    // [M]
  float* bal = ent + M_;         // [M]

  // bf16 weight cache (converted once per launch, reused by all 3 iterations)
  const long EE = (long)E_ * E_;
  __bf16* wb   = (__bf16*)(ws + 16 * ME_ + 2 * (long)M_);
  __bf16* w_sr = wb;            __bf16* w_q  = wb + 1 * EE;  __bf16* w_k  = wb + 2 * EE;
  __bf16* w_v  = wb + 3 * EE;   __bf16* w_o  = wb + 4 * EE;  __bf16* w_c  = wb + 5 * EE;
  __bf16* w_qp = wb + 6 * EE;   __bf16* w_kp = wb + 7 * EE;  __bf16* w_ob = wb + 8 * EE;
  __bf16* w_ig = wb + 9 * EE;   /* 2*EE */
  __bf16* w_f1 = wb + 11 * EE;  /* 4*EE */
  __bf16* w_f2 = wb + 15 * EE;  /* 4*EE */

  float* xc = x0;
  float* xo = x1;

  const dim3 T(256);
  const dim3 GE(M_ / 128, E_ / 128, 1);    // token x E GEMMs
  const dim3 GF(M_ / 128, F_ / 128, 1);    // FFN up
  const dim3 GS(S_ / 128, S_ / 128, B_);   // batched sim
  const dim3 GT(S_ / 128, E_ / 128, B_);   // batched transfer
  const dim3 GROW(M_);
  const dim3 GW(EE / 1024);                // E*E convert

  // --- one-time (per launch) weight f32 -> bf16 conversion ---
  cvt_bf16_k<<<GW, T, 0, stream>>>(sr_w, w_sr);
  cvt_bf16_k<<<GW, T, 0, stream>>>(q_w, w_q);
  cvt_bf16_k<<<GW, T, 0, stream>>>(k_w, w_k);
  cvt_bf16_k<<<GW, T, 0, stream>>>(v_w, w_v);
  cvt_bf16_k<<<GW, T, 0, stream>>>(o_w, w_o);
  cvt_bf16_k<<<GW, T, 0, stream>>>(c_w, w_c);
  cvt_bf16_k<<<GW, T, 0, stream>>>(qp_w, w_qp);
  cvt_bf16_k<<<GW, T, 0, stream>>>(kp_w, w_kp);
  cvt_bf16_k<<<GW, T, 0, stream>>>(obs_w, w_ob);
  cvt_bf16_k<<<dim3(2 * EE / 1024), T, 0, stream>>>(ig_w, w_ig);
  cvt_bf16_k<<<dim3(4 * EE / 1024), T, 0, stream>>>(f1_w, w_f1);
  cvt_bf16_k<<<dim3(4 * EE / 1024), T, 0, stream>>>(f2_w, w_f2);

  init_xk_k<<<GROW, T, 0, stream>>>(x_in, xc, ksb);

  for (int it = 0; it < 3; ++it) {
    // --- self-referential attention ---
    layernorm_k<<<GROW, T, 0, stream>>>(xc, ln1_g, ln1_b, xn);
    gemm_wmma_k<EPI_RES_SCALE, false, true><<<GE, T, 0, stream>>>(xn, w_sr, sr_b, xn, refg, xr,
                                                                  E_, E_, E_, E_, 0, 0, 0);
    gemm_wmma_k<EPI_BIAS, false, true><<<GE, T, 0, stream>>>(xr, w_q, q_b, nullptr, nullptr, qb,
                                                             E_, E_, E_, E_, 0, 0, 0);
    gemm_wmma_k<EPI_BIAS, false, true><<<GE, T, 0, stream>>>(xr, w_k, k_b, nullptr, nullptr, kb,
                                                             E_, E_, E_, E_, 0, 0, 0);
    gemm_wmma_k<EPI_BIAS, false, true><<<GE, T, 0, stream>>>(xn, w_v, v_b, nullptr, nullptr, vb,
                                                             E_, E_, E_, E_, 0, 0, 0);
    flash_attn_k<<<dim3(S_ / 128, H_, B_), T, 0, stream>>>(qb, kb, vb, ao);
    gemm_wmma_k<EPI_BIAS, false, true><<<GE, T, 0, stream>>>(ao, w_o, o_b, nullptr, nullptr, qu,
                                                             E_, E_, E_, E_, 0, 0, 0);
    // --- FFN (xn2 reuses xr) ---
    layernorm_k<<<GROW, T, 0, stream>>>(xc, ln2_g, ln2_b, xr);
    gemm_wmma_k<EPI_GELU, false, true><<<GF, T, 0, stream>>>(xr, w_f1, f1_b, nullptr, nullptr, hb,
                                                             E_, E_, F_, F_, 0, 0, 0);
    gemm_wmma_k<EPI_BIAS, false, true><<<GE, T, 0, stream>>>(hb, w_f2, f2_b, nullptr, nullptr, cl,
                                                             F_, F_, E_, E_, 0, 0, 0);
    // --- quantum-classical interface ---
    gemm_wmma_k<EPI_BIAS, false, true><<<GE, T, 0, stream>>>(cl, w_c, c_b, nullptr, nullptr, qb,
                                                             E_, E_, E_, E_, 0, 0, 0);
    gemm_wmma_k<EPI_BIAS, false, true><<<GE, T, 0, stream>>>(qu, w_qp, qp_b, nullptr, nullptr, kb,
                                                             E_, E_, E_, E_, 0, 0, 0);
    // gate = sigmoid(c_proj@W1 + q_proj@W2 + b), split concat GEMM
    gemm_wmma_k<EPI_BIAS, false, true><<<GE, T, 0, stream>>>(qb, w_ig, ig_b, nullptr, nullptr, ao,
                                                             E_, E_, E_, E_, 0, 0, 0);
    gemm_wmma_k<EPI_ADD_SIGMOID, false, true><<<GE, T, 0, stream>>>(kb, w_ig + EE, nullptr,
                                                                    nullptr, nullptr, ao,
                                                                    E_, E_, E_, E_, 0, 0, 0);
    // sim = c_proj @ q_proj^T, batched over b (both operands are live f32 activations)
    gemm_wmma_k<EPI_BIAS, true, false><<<GS, T, 0, stream>>>(qb, kb, nullptr, nullptr, nullptr, sim,
                                                             E_, E_, E_, S_,
                                                             (long)S_ * E_, (long)S_ * E_, (long)S_ * S_);
    softmax_rows_k<<<dim3(B_ * S_), T, 0, stream>>>(sim, temp);
    gemm_wmma_k<EPI_BIAS, false, false><<<GT, T, 0, stream>>>(sim, qu, nullptr, nullptr, nullptr, vb,
                                                              S_, S_, E_, E_,
                                                              (long)S_ * S_, (long)S_ * E_, (long)S_ * E_);
    iface_k<<<GROW, T, 0, stream>>>(cl, ao, vb);       // iface = classical + gate*transfer
    // --- entropy regulator ---
    ent_bal_k<<<GROW, T, 0, stream>>>(cl, ee_w, ee_b, ent, bal);
    gemm_wmma_k<EPI_BIAS, false, true><<<GE, T, 0, stream>>>(ksb, w_kp, kp_b, nullptr, nullptr, xn,
                                                             E_, E_, E_, E_, 0, 0, 0);
    reg_k<<<GROW, T, 0, stream>>>(xc, cl, xn, ent, bal, eta);   // x += regulated
    // --- observer feedback: x_next = x + obs_gate*(x@obs_w + obs_b) ---
    gemm_wmma_k<EPI_RES_SCALE, false, true><<<GE, T, 0, stream>>>(xc, w_ob, obs_b, xc, obsg, xo,
                                                                  E_, E_, E_, E_, 0, 0, 0);
    ks_upd_k<<<GROW, T, 0, stream>>>(ksb, xo);
    float* t = xc; xc = xo; xo = t;
  }

  float* out = (float*)d_out;                 // tuple: [x (ME_), ks (ME_)]
  layernorm_k<<<GROW, T, 0, stream>>>(xc, ln3_g, ln3_b, out);
  copy_k<<<GROW, T, 0, stream>>>(ksb, out + ME_);
}
